// FeatureExtraction_7705171329304
// MI455X (gfx1250) — matrix operand order repaired and checked by reading, not compiled, not run on hardware
//
#include <hip/hip_runtime.h>
#include <hip/hip_bf16.h>
#include <math.h>

typedef __attribute__((ext_vector_type(16))) _Float16 v16h;
typedef __attribute__((ext_vector_type(8)))  _Float16 v8h;
typedef __attribute__((ext_vector_type(8)))  float    v8f;

// ---------------------------------------------------------------------------
// WMMA fragment helpers (layouts per CDNA5 ISA 7.12.2, wave32)
// ---------------------------------------------------------------------------

// 16-bit A matrix 16x32: lane m=lane&15, half=lane>>4.
// elems 0..7  -> k = k0 + half*8 + i        (contiguous 8)
// elems 8..15 -> k = k0 + 16 + half*8 + i-8 (contiguous 8)
static __device__ __forceinline__ v16h load_frag_a(const _Float16* rowp, int k0, int half) {
  const _Float16* p = rowp + k0 + half * 8;
  v8h lo = *(const v8h*)(p);
  v8h hi = *(const v8h*)(p + 16);
  return __builtin_shufflevector(lo, hi, 0,1,2,3,4,5,6,7,8,9,10,11,12,13,14,15);
}

// B matrix 32x16 stored as pre-transposed weights WT[N][Kp] (row-major, N-major):
// lane n=lane&15 holds column n0+n, elems 0..7 -> k = k0 + half*8 + i, elems 8..15 -> k = k0 + 16 + half*8 + i-8
static __device__ __forceinline__ v16h load_frag_b(const _Float16* WT, int Kp, int n0, int k0, int lane) {
  const int n = lane & 15, half = lane >> 4;
  const _Float16* p = WT + (size_t)(n0 + n) * Kp + k0 + half * 8;
  v8h lo = *(const v8h*)(p);
  v8h hi = *(const v8h*)(p + 16);
  return __builtin_shufflevector(lo, hi, 0,1,2,3,4,5,6,7,8,9,10,11,12,13,14,15);
}

static __device__ __forceinline__ v8f wmma_f16(v16h a, v16h b, v8f c) {
  // 8 args: (neg_a, A, neg_b, B, c_mod, C, reuse_a, reuse_b)
  return __builtin_amdgcn_wmma_f32_16x16x32_f16(false, a, false, b, (short)0, c, false, false);
}

// ---------------------------------------------------------------------------
// Tiny scalar kernels
// ---------------------------------------------------------------------------

__global__ void calc_masks_kernel(const float* __restrict__ rho, int* __restrict__ m3,
                                  int* __restrict__ m4, int B) {
  int b = threadIdx.x;
  if (b >= B) return;
  float rs = rho[b] * 4.0f;
  float nl = ceilf(4.0f - 3.0f * logf(0.396f * rs + 1.0f));
  m4[b] = (nl > 3.5f) ? 1 : 0;
  m3[b] = (nl > 2.5f) ? 1 : 0;
}

// f32 weight (K x N row-major) -> f16 transposed & zero-padded WT (Np x Kp row-major)
__global__ void cvt_wt_kernel(const float* __restrict__ src, _Float16* __restrict__ dst,
                              int K, int N, int Kp, int Np) {
  int idx = blockIdx.x * blockDim.x + threadIdx.x;
  if (idx >= Kp * Np) return;
  int n = idx / Kp, k = idx % Kp;
  float v = (k < K && n < N) ? src[(size_t)k * N + n] : 0.0f;
  dst[idx] = (_Float16)v;
}

// h0 = relu(x @ w_start + b_start): K=3 -> plain VALU, one thread per point
__global__ void start_layer_kernel(const float* __restrict__ x, const float* __restrict__ w,
                                   const float* __restrict__ b, _Float16* __restrict__ H0,
                                   int rows) {
  int r = blockIdx.x * blockDim.x + threadIdx.x;
  if (r >= rows) return;
  float x0 = x[(size_t)r * 3 + 0];
  float x1 = x[(size_t)r * 3 + 1];
  float x2 = x[(size_t)r * 3 + 2];
  _Float16* o = H0 + (size_t)r * 32;
  #pragma unroll 8
  for (int c = 0; c < 32; ++c) {
    float v = x0 * w[c] + x1 * w[32 + c] + x2 * w[64 + c] + b[c];
    o[c] = (_Float16)(v > 0.0f ? v : 0.0f);
  }
}

// grouped max-pool by stride s along the point axis (values are post-relu >= 0)
__global__ void maxpool_kernel(const _Float16* __restrict__ in, _Float16* __restrict__ out,
                               int orows, int Cp, int s) {
  int idx = blockIdx.x * blockDim.x + threadIdx.x;
  if (idx >= orows * Cp) return;
  int r = idx / Cp, c = idx % Cp;
  const _Float16* p = in + (size_t)r * s * Cp + c;
  float mx = (float)p[0];
  for (int j = 1; j < s; ++j) mx = fmaxf(mx, (float)p[(size_t)j * Cp]);
  out[idx] = (_Float16)mx;
}

// where(mask[b], dst, skip) -> overwrite rows of batches with mask==0
__global__ void merge_skip_kernel(_Float16* __restrict__ dst, const _Float16* __restrict__ skip,
                                  const int* __restrict__ mask, int rows_per_batch, int Cp,
                                  long total) {
  long idx = (long)blockIdx.x * blockDim.x + threadIdx.x;
  if (idx >= total) return;
  int b = (int)(idx / ((long)rows_per_batch * Cp));
  if (!mask[b]) dst[idx] = skip[idx];
}

// ---------------------------------------------------------------------------
// Generic WMMA GEMM: Out[rows, Np] = relu( up(A1, s1) @ W1 (+ A2 @ W2) + bias )
// One wave per 16-row tile, 4 waves / 128-thread block.
// ---------------------------------------------------------------------------
__global__ void __launch_bounds__(128)
gemm_wmma_kernel(const _Float16* __restrict__ A1, int K1p, int s1,
                 const _Float16* __restrict__ A2, int K2p,
                 const _Float16* __restrict__ W1T, const _Float16* __restrict__ W2T,
                 const float* __restrict__ bias, int realN,
                 _Float16* __restrict__ Out, int Np, int rows) {
  const int lane = threadIdx.x & 31;
  const int wave = threadIdx.x >> 5;
  const int row0 = (blockIdx.x * 4 + wave) * 16;
  if (row0 >= rows) return;
  const int m = lane & 15, half = lane >> 4;
  const _Float16* a1row = A1 + (size_t)((row0 + m) / s1) * K1p;
  const _Float16* a2row = A2 ? (A2 + (size_t)(row0 + m) * K2p) : nullptr;

  for (int n0 = 0; n0 < Np; n0 += 16) {
    v8f c = {0.f, 0.f, 0.f, 0.f, 0.f, 0.f, 0.f, 0.f};
    for (int k0 = 0; k0 < K1p; k0 += 32)
      c = wmma_f16(load_frag_a(a1row, k0, half), load_frag_b(W1T, K1p, n0, k0, lane), c);
    if (a2row)
      for (int k0 = 0; k0 < K2p; k0 += 32)
        c = wmma_f16(load_frag_a(a2row, k0, half), load_frag_b(W2T, K2p, n0, k0, lane), c);
    const int n = n0 + m;
    const float bv = (n < realN) ? bias[n] : 0.0f;  // padded cols stay exactly 0
    #pragma unroll
    for (int r = 0; r < 8; ++r) {
      float v = c[r] + bv;
      v = v > 0.0f ? v : 0.0f;
      Out[(size_t)(row0 + half * 8 + r) * Np + n] = (_Float16)v;
    }
  }
}

// ---------------------------------------------------------------------------
// Fused head: tanh(relu(relu(d3 @ w1) @ w2 + b2) @ w3 + b3)
// Keeps the 16x128 / 16x64 intermediates in LDS (D-fragment -> A-fragment relayout)
// ---------------------------------------------------------------------------
__global__ void __launch_bounds__(128)
head_wmma_kernel(const _Float16* __restrict__ D3,
                 const _Float16* __restrict__ W1T, const _Float16* __restrict__ W2T,
                 const float* __restrict__ b2, const float* __restrict__ w3,
                 const float* __restrict__ b3, float* __restrict__ out, int rows) {
  __shared__ _Float16 smem[4][16 * 128 + 16 * 64];
  const int lane = threadIdx.x & 31;
  const int wave = threadIdx.x >> 5;
  const int row0 = (blockIdx.x * 4 + wave) * 16;  // grid is exact; no early return (barriers)
  const int m = lane & 15, half = lane >> 4;
  _Float16* y1 = smem[wave];
  _Float16* y2 = smem[wave] + 16 * 128;

  // stage 1: y1 = relu(d3 @ w1), K=32 (single k-step), N=128
  {
    const _Float16* arow = D3 + (size_t)(row0 + m) * 32;
    v16h a = load_frag_a(arow, 0, half);
    for (int n0 = 0; n0 < 128; n0 += 16) {
      v8f c = {0.f, 0.f, 0.f, 0.f, 0.f, 0.f, 0.f, 0.f};
      c = wmma_f16(a, load_frag_b(W1T, 32, n0, 0, lane), c);
      const int n = n0 + m;
      #pragma unroll
      for (int r = 0; r < 8; ++r) {
        float v = c[r];
        y1[(size_t)(half * 8 + r) * 128 + n] = (_Float16)(v > 0.0f ? v : 0.0f);
      }
    }
  }
  __syncthreads();

  // stage 2: y2 = relu(y1 @ w2 + b2), K=128, N=64
  {
    const _Float16* arow = y1 + (size_t)m * 128;
    for (int n0 = 0; n0 < 64; n0 += 16) {
      v8f c = {0.f, 0.f, 0.f, 0.f, 0.f, 0.f, 0.f, 0.f};
      #pragma unroll
      for (int k0 = 0; k0 < 128; k0 += 32)
        c = wmma_f16(load_frag_a(arow, k0, half), load_frag_b(W2T, 128, n0, k0, lane), c);
      const int n = n0 + m;
      const float bv = b2[n];
      #pragma unroll
      for (int r = 0; r < 8; ++r) {
        float v = c[r] + bv;
        y2[(size_t)(half * 8 + r) * 64 + n] = (_Float16)(v > 0.0f ? v : 0.0f);
      }
    }
  }
  __syncthreads();

  // stage 3: out = tanh(y2 @ w3 + b3), K=64, N=3 -> VALU, lanes 0..15 take one row each
  if (lane < 16) {
    const _Float16* yr = y2 + (size_t)lane * 64;
    float a0 = b3[0], a1 = b3[1], a2 = b3[2];
    #pragma unroll 8
    for (int k = 0; k < 64; ++k) {
      float v = (float)yr[k];
      a0 += v * w3[k * 3 + 0];
      a1 += v * w3[k * 3 + 1];
      a2 += v * w3[k * 3 + 2];
    }
    size_t ro = (size_t)(row0 + lane) * 3;
    out[ro + 0] = tanhf(a0);
    out[ro + 1] = tanhf(a1);
    out[ro + 2] = tanhf(a2);
  }
}

// ---------------------------------------------------------------------------
// Host launcher
// ---------------------------------------------------------------------------
extern "C" void kernel_launch(void* const* d_in, const int* in_sizes, int n_in,
                              void* d_out, int out_size, void* d_ws, size_t ws_size,
                              hipStream_t stream) {
  (void)in_sizes; (void)n_in; (void)out_size; (void)ws_size;

  constexpr int B = 16, N = 24576;
  constexpr int R0 = B * N;            // 393216
  constexpr int R1 = R0 / 4;           // 98304
  constexpr int R2 = R1 / 3;           // 32768
  constexpr int R3 = R2 / 2;           // 16384
  // padded channel widths (multiples of 32)
  constexpr int C0 = 32, C1 = 64, C2 = 96, C3 = 128, C4 = 192, CH1 = 128, CH2 = 64;

  // --- inputs (setup_inputs dict order, lists flattened in order) ---
  const float* x       = (const float*)d_in[1];
  const float* rho     = (const float*)d_in[3];
  const float* w_start = (const float*)d_in[4];
  const float* b_start = (const float*)d_in[5];
  const float* w_e[4]  = {(const float*)d_in[6],  (const float*)d_in[7],
                          (const float*)d_in[8],  (const float*)d_in[9]};
  const float* b_e[4]  = {(const float*)d_in[10], (const float*)d_in[11],
                          (const float*)d_in[12], (const float*)d_in[13]};
  const float* w_up[4] = {(const float*)d_in[14], (const float*)d_in[15],
                          (const float*)d_in[16], (const float*)d_in[17]};
  const float* w_sk[4] = {(const float*)d_in[18], (const float*)d_in[19],
                          (const float*)d_in[20], (const float*)d_in[21]};
  const float* b_d[4]  = {(const float*)d_in[22], (const float*)d_in[23],
                          (const float*)d_in[24], (const float*)d_in[25]};
  const float* w1 = (const float*)d_in[26];
  const float* w2 = (const float*)d_in[27];
  const float* b2 = (const float*)d_in[28];
  const float* w3 = (const float*)d_in[29];
  const float* b3 = (const float*)d_in[30];
  float* out = (float*)d_out;

  // --- workspace carve-out (256B aligned) ---
  char* base = (char*)d_ws;
  size_t off = 0;
  auto alloc = [&](size_t bytes) -> void* {
    void* p = (void*)(base + off);
    off = (off + bytes + 255) & ~(size_t)255;
    return p;
  };
  int* m3 = (int*)alloc(B * sizeof(int));
  int* m4 = (int*)alloc(B * sizeof(int));
  auto hf = [&](size_t elems) -> _Float16* { return (_Float16*)alloc(elems * sizeof(_Float16)); };

  // transposed/padded f16 weights: WT[Np][Kp]
  _Float16* WT_e0  = hf((size_t)C1 * C0);   // 64x32
  _Float16* WT_e1  = hf((size_t)C2 * C1);   // 96x64
  _Float16* WT_e2  = hf((size_t)C3 * C2);   // 128x96
  _Float16* WT_e3  = hf((size_t)C4 * C3);   // 192x128
  _Float16* WT_up0 = hf((size_t)C3 * C4);   // 128x192
  _Float16* WT_up1 = hf((size_t)C2 * C3);   // 96x128
  _Float16* WT_up2 = hf((size_t)C1 * C2);   // 64x96
  _Float16* WT_up3 = hf((size_t)C0 * C1);   // 32x64
  _Float16* WT_sk0 = hf((size_t)C3 * C3);   // 128x128
  _Float16* WT_sk1 = hf((size_t)C2 * C2);   // 96x96
  _Float16* WT_sk2 = hf((size_t)C1 * C1);   // 64x64
  _Float16* WT_sk3 = hf((size_t)C0 * C0);   // 32x32
  _Float16* WT_w1  = hf((size_t)CH1 * C0);  // 128x32
  _Float16* WT_w2  = hf((size_t)CH2 * CH1); // 64x128

  // padded f16 activations
  _Float16* H0  = hf((size_t)R0 * C0);
  _Float16* P0  = hf((size_t)R1 * C0);
  _Float16* H1  = hf((size_t)R1 * C1);
  _Float16* P1  = hf((size_t)R2 * C1);
  _Float16* H2  = hf((size_t)R2 * C2);
  _Float16* P2  = hf((size_t)R3 * C2);
  _Float16* H3  = hf((size_t)R3 * C3);
  _Float16* H4  = hf((size_t)R3 * C4);
  _Float16* D0M = hf((size_t)R3 * C3);
  _Float16* D1M = hf((size_t)R2 * C2);
  _Float16* D2  = hf((size_t)R1 * C1);
  _Float16* D3  = hf((size_t)R0 * C0);

  const _Float16* NOA = nullptr;

  // --- per-batch adaptive-depth masks ---
  calc_masks_kernel<<<1, 32, 0, stream>>>(rho, m3, m4, B);

  // --- weight conversion (f32 KxN -> f16 NpxKp transposed, zero-padded) ---
  auto cvt = [&](const float* src, _Float16* dst, int K, int Nn, int Kp, int Np) {
    int total = Kp * Np;
    cvt_wt_kernel<<<(total + 255) / 256, 256, 0, stream>>>(src, dst, K, Nn, Kp, Np);
  };
  cvt(w_e[0],  WT_e0,  32, 48,  C0, C1);
  cvt(w_e[1],  WT_e1,  48, 72,  C1, C2);
  cvt(w_e[2],  WT_e2,  72, 108, C2, C3);
  cvt(w_e[3],  WT_e3,  108, 162, C3, C4);
  cvt(w_up[0], WT_up0, 162, 108, C4, C3);
  cvt(w_up[1], WT_up1, 108, 72,  C3, C2);
  cvt(w_up[2], WT_up2, 72, 48,   C2, C1);
  cvt(w_up[3], WT_up3, 48, 32,   C1, C0);
  cvt(w_sk[0], WT_sk0, 108, 108, C3, C3);
  cvt(w_sk[1], WT_sk1, 72, 72,   C2, C2);
  cvt(w_sk[2], WT_sk2, 48, 48,   C1, C1);
  cvt(w_sk[3], WT_sk3, 32, 32,   C0, C0);
  cvt(w1,      WT_w1,  32, 128,  C0, CH1);
  cvt(w2,      WT_w2,  128, 64,  CH1, CH2);

  auto gemm = [&](const _Float16* A1, int K1p, int s1, const _Float16* A2, int K2p,
                  const _Float16* W1T, const _Float16* W2T, const float* bias, int realN,
                  _Float16* Out, int Np, int rows) {
    gemm_wmma_kernel<<<rows / 64, 128, 0, stream>>>(A1, K1p, s1, A2, K2p, W1T, W2T,
                                                    bias, realN, Out, Np, rows);
  };
  auto pool = [&](const _Float16* in, _Float16* o, int orows, int Cp, int s) {
    int total = orows * Cp;
    maxpool_kernel<<<(total + 255) / 256, 256, 0, stream>>>(in, o, orows, Cp, s);
  };

  // --- encoder ---
  start_layer_kernel<<<(R0 + 255) / 256, 256, 0, stream>>>(x, w_start, b_start, H0, R0);
  pool(H0, P0, R1, C0, 4);
  gemm(P0, C0, 1, NOA, 0, WT_e0, nullptr, b_e[0], 48,  H1, C1, R1);
  pool(H1, P1, R2, C1, 3);
  gemm(P1, C1, 1, NOA, 0, WT_e1, nullptr, b_e[1], 72,  H2, C2, R2);
  pool(H2, P2, R3, C2, 2);
  gemm(P2, C2, 1, NOA, 0, WT_e2, nullptr, b_e[2], 108, H3, C3, R3);
  gemm(H3, C3, 1, NOA, 0, WT_e3, nullptr, b_e[3], 162, H4, C4, R3);  // pool stride 1

  // --- adaptive-depth decoder ---
  // d0 = relu(up(H4,1)@w_up0 + H3@w_sk0 + b_d0); h3 = where(m4, d0, H3)
  gemm(H4, C4, 1, H3, C3, WT_up0, WT_sk0, b_d[0], 108, D0M, C3, R3);
  merge_skip_kernel<<<((long)R3 * C3 + 255) / 256, 256, 0, stream>>>(
      D0M, H3, m4, N / 24, C3, (long)R3 * C3);
  // d1 = relu(up(h3,2)@w_up1 + H2@w_sk1 + b_d1); h2 = where(m3, d1, H2)
  gemm(D0M, C3, 2, H2, C2, WT_up1, WT_sk1, b_d[1], 72, D1M, C2, R2);
  merge_skip_kernel<<<((long)R2 * C2 + 255) / 256, 256, 0, stream>>>(
      D1M, H2, m3, N / 12, C2, (long)R2 * C2);
  // d2 = relu(up(h2,3)@w_up2 + H1@w_sk2 + b_d2)
  gemm(D1M, C2, 3, H1, C1, WT_up2, WT_sk2, b_d[2], 48, D2, C1, R1);
  // d3 = relu(up(d2,4)@w_up3 + H0@w_sk3 + b_d3)
  gemm(D2, C1, 4, H0, C0, WT_up3, WT_sk3, b_d[3], 32, D3, C0, R0);

  // --- fused classification head ---
  head_wmma_kernel<<<R0 / 64, 128, 0, stream>>>(D3, WT_w1, WT_w2, b2, w3, b3, out, R0);
}
